// WindowAttention_67963562492510
// MI455X (gfx1250) — compile-verified, matrix operand order checked by reading, not run on hardware
//
#include <hip/hip_runtime.h>

#define NUM_HEADS 4
#define NDIM 128
#define NTOK 64
#define NWIN 4096
#define NB   8192
#define HD   32

typedef __attribute__((ext_vector_type(16))) _Float16 v16h;
typedef __attribute__((ext_vector_type(8)))  _Float16 v8h;
typedef __attribute__((ext_vector_type(4)))  _Float16 v4h;
typedef __attribute__((ext_vector_type(8)))  float    v8f;

// LDS strides (in halfs), padded so 16 lanes doing b128 reads spread across banks
#define XS 136      // x / mask / attn-out buffer [64][128]
#define QS 264      // q|k buffer                 [64][256]
#define VS 72       // v^T                        [4*32][64]
#define PS 72       // probs                      [4*64][64]

#define LDS_X  0
#define LDS_QK 17408     // 64*136*2
#define LDS_VT 51200     // + 64*264*2
#define LDS_PR 69632     // + 128*72*2
#define LDS_BYTES 106496 // + 256*72*2

// ws layout (bytes)
#define WS_WT   0        // qkv_w transposed, f16 [384][128]
#define WS_PT   98304    // proj_w transposed, f16 [128][128]
#define WS_BIAS 131072   // rel bias, f32 [4][64][64]

__device__ __forceinline__ v16h frag_ld(const _Float16* base, int r, int half,
                                        int stride, int k0) {
  // A-style fragment (also B via K^T/V^T layouts): lane r<16 -> row r,
  // elems 0-7 = K k0..k0+7 (+half*8), elems 8-15 = K +16
  const _Float16* p = base + r * stride + k0 + half * 8;
  v8h lo = *(const v8h*)(p);
  v8h hi = *(const v8h*)(p + 16);
  return __builtin_shufflevector(lo, hi, 0,1,2,3,4,5,6,7,8,9,10,11,12,13,14,15);
}

__device__ __forceinline__ v8f wmma16(v16h a, v16h b, v8f c) {
  return __builtin_amdgcn_wmma_f32_16x16x32_f16(false, a, false, b, (short)0, c,
                                                false, false);
}

__global__ void wa_prep(const float* __restrict__ qkv_w,
                        const float* __restrict__ proj_w,
                        const float* __restrict__ rel_table,
                        const int*   __restrict__ rel_index,
                        _Float16* __restrict__ wT, _Float16* __restrict__ pT,
                        float* __restrict__ biasw) {
  int i = blockIdx.x * 256 + threadIdx.x;
  if (i < 384 * 128) {                      // wT[n][k] = qkv_w[k][n]
    int n = i >> 7, k = i & 127;
    wT[i] = (_Float16)qkv_w[k * 384 + n];
    return;
  }
  int j = i - 384 * 128;
  if (j < 128 * 128) {                      // pT[n][k] = proj_w[k][n]
    int n = j >> 7, k = j & 127;
    pT[j] = (_Float16)proj_w[k * 128 + n];
    return;
  }
  int l = j - 128 * 128;
  if (l < NUM_HEADS * NTOK * NTOK) {        // biasw[h][m][n]
    int h = l >> 12, mn = l & 4095;
    biasw[l] = rel_table[rel_index[mn] * 4 + h];
  }
}

__launch_bounds__(256)
__global__ void wa_main(const float* __restrict__ x, const float* __restrict__ mask,
                        const float* __restrict__ qkv_b, const float* __restrict__ proj_b,
                        const _Float16* __restrict__ wT, const _Float16* __restrict__ pT,
                        const float* __restrict__ biasw, float* __restrict__ out) {
  __shared__ __align__(16) unsigned char smem[LDS_BYTES];
  _Float16* xl = (_Float16*)(smem + LDS_X);
  _Float16* qk = (_Float16*)(smem + LDS_QK);
  _Float16* vt = (_Float16*)(smem + LDS_VT);
  _Float16* pr = (_Float16*)(smem + LDS_PR);
  float*    ml = (float*)(smem + LDS_X);   // mask overlay (x dead after QKV)
  _Float16* ol = xl;                       // attn-out overlay (mask dead after softmax)

  const int tid   = threadIdx.x;
  const int lane  = tid & 31;
  const int wid   = tid >> 5;     // 8 waves
  const int col16 = lane & 15;
  const int half  = lane >> 4;
  const int b     = blockIdx.x;
  const int win   = b & (NWIN - 1);
  const float scale = 0.1767766952966369f;  // 1/sqrt(32)

  const float* xb = x + (size_t)b * (NTOK * NDIM);
  const float* mb = mask + (size_t)win * (NTOK * NTOK);
  __builtin_prefetch(mb + tid * 16, 0, 0);  // warm this window's 16KB mask

  // ---- stage 0: x -> f16 LDS (vectorized b128 loads, b64 stores) ----
  {
    const float4* xb4 = (const float4*)xb;
#pragma unroll
    for (int it = 0; it < 8; ++it) {
      int i = tid + it * 256;                 // 2048 float4 total
      float4 v = xb4[i];
      int r = i >> 5, c = (i & 31) * 4;
      v4h h = {(_Float16)v.x, (_Float16)v.y, (_Float16)v.z, (_Float16)v.w};
      *(v4h*)(xl + r * XS + c) = h;
    }
  }
  __syncthreads();

  // ---- stage 1: qkv = x @ qkv_w + b (wave: 1 m-tile x 12 n-tiles) ----
  {
    const int m0 = (wid >> 1) * 16;
    v16h a0 = frag_ld(xl + m0 * XS, col16, half, XS, 0);
    v16h a1 = frag_ld(xl + m0 * XS, col16, half, XS, 32);
    v16h a2 = frag_ld(xl + m0 * XS, col16, half, XS, 64);
    v16h a3 = frag_ld(xl + m0 * XS, col16, half, XS, 96);
    const int ntbase = (wid & 1) * 12;
    for (int t = 0; t < 12; ++t) {
      int n0 = (ntbase + t) * 16;
      const _Float16* wrow = wT + (size_t)n0 * NDIM;
      v8f acc = {};
      acc = wmma16(a0, frag_ld(wrow, col16, half, NDIM, 0),  acc);
      acc = wmma16(a1, frag_ld(wrow, col16, half, NDIM, 32), acc);
      acc = wmma16(a2, frag_ld(wrow, col16, half, NDIM, 64), acc);
      acc = wmma16(a3, frag_ld(wrow, col16, half, NDIM, 96), acc);
      int coln = n0 + col16;
      float bv = qkv_b[coln];
#pragma unroll
      for (int r = 0; r < 8; ++r) {
        int row = m0 + half * 8 + r;
        float v = acc[r] + bv;
        if (coln < 128) {                       // Q (pre-scaled)
          qk[row * QS + coln] = (_Float16)(v * scale);
        } else if (coln < 256) {                // K
          qk[row * QS + coln] = (_Float16)v;
        } else {                                // V -> transposed [h*32+d][token]
          vt[(coln - 256) * VS + row] = (_Float16)v;
        }
      }
    }
  }
  __syncthreads();

  // ---- stage 2: stage mask->LDS, S = Q K^T, softmax(+bias+mask), probs->LDS ----
  {
    // cooperative mask copy into the (now dead) x buffer: 4 x b128 per thread,
    // overlapped with the score WMMAs below, made visible by the barrier.
    const float4* mb4 = (const float4*)mb;
#pragma unroll
    for (int it = 0; it < 4; ++it) {
      int i = tid + it * 256;                  // 1024 float4 total
      ((float4*)ml)[i] = mb4[i];
    }

    const int h  = wid >> 1;
    const int m0 = (wid & 1) * 32;          // wave owns 32 full rows of head h
    v16h q0 = frag_ld(qk + (m0)      * QS + h * HD, col16, half, QS, 0);
    v16h q1 = frag_ld(qk + (m0 + 16) * QS + h * HD, col16, half, QS, 0);
    v16h bk0 = frag_ld(qk + 0  * QS + 128 + h * HD, col16, half, QS, 0);
    v16h bk1 = frag_ld(qk + 16 * QS + 128 + h * HD, col16, half, QS, 0);
    v16h bk2 = frag_ld(qk + 32 * QS + 128 + h * HD, col16, half, QS, 0);
    v16h bk3 = frag_ld(qk + 48 * QS + 128 + h * HD, col16, half, QS, 0);
    v8f z = {};
    v8f s[2][4];
    s[0][0] = wmma16(q0, bk0, z); s[0][1] = wmma16(q0, bk1, z);
    s[0][2] = wmma16(q0, bk2, z); s[0][3] = wmma16(q0, bk3, z);
    s[1][0] = wmma16(q1, bk0, z); s[1][1] = wmma16(q1, bk1, z);
    s[1][2] = wmma16(q1, bk2, z); s[1][3] = wmma16(q1, bk3, z);

    __syncthreads();   // mask copy visible; qk reads done

    const float* bh = biasw + h * (NTOK * NTOK);
#pragma unroll
    for (int mt2 = 0; mt2 < 2; ++mt2) {
#pragma unroll
      for (int r = 0; r < 8; ++r) {
        int M = m0 + mt2 * 16 + half * 8 + r;
        float v0 = s[mt2][0][r] + bh[M * 64 +  0 + col16] + ml[M * 64 +  0 + col16];
        float v1 = s[mt2][1][r] + bh[M * 64 + 16 + col16] + ml[M * 64 + 16 + col16];
        float v2 = s[mt2][2][r] + bh[M * 64 + 32 + col16] + ml[M * 64 + 32 + col16];
        float v3 = s[mt2][3][r] + bh[M * 64 + 48 + col16] + ml[M * 64 + 48 + col16];
        float mx = fmaxf(fmaxf(v0, v1), fmaxf(v2, v3));
#pragma unroll
        for (int d = 1; d < 16; d <<= 1) mx = fmaxf(mx, __shfl_xor(mx, d, 32));
        float e0 = __expf(v0 - mx), e1 = __expf(v1 - mx);
        float e2 = __expf(v2 - mx), e3 = __expf(v3 - mx);
        float sm = (e0 + e1) + (e2 + e3);
#pragma unroll
        for (int d = 1; d < 16; d <<= 1) sm += __shfl_xor(sm, d, 32);
        float inv = 1.0f / sm;
        _Float16* prow = pr + (size_t)(h * 64 + M) * PS;
        prow[ 0 + col16] = (_Float16)(e0 * inv);
        prow[16 + col16] = (_Float16)(e1 * inv);
        prow[32 + col16] = (_Float16)(e2 * inv);
        prow[48 + col16] = (_Float16)(e3 * inv);
      }
    }
  }
  __syncthreads();

  // ---- stage 3: O_h = P V (wave: head h, 2 m-tiles x 2 n-tiles) ----
  {
    const int h  = wid & 3;
    const int mg = wid >> 2;
    const _Float16* pb = pr + (size_t)h * 64 * PS;
    const _Float16* vb = vt + (size_t)h * HD * VS;
    v16h p00 = frag_ld(pb + (mg * 32)      * PS, col16, half, PS, 0);
    v16h p01 = frag_ld(pb + (mg * 32)      * PS, col16, half, PS, 32);
    v16h p10 = frag_ld(pb + (mg * 32 + 16) * PS, col16, half, PS, 0);
    v16h p11 = frag_ld(pb + (mg * 32 + 16) * PS, col16, half, PS, 32);
    v16h bv00 = frag_ld(vb,           col16, half, VS, 0);
    v16h bv01 = frag_ld(vb,           col16, half, VS, 32);
    v16h bv10 = frag_ld(vb + 16 * VS, col16, half, VS, 0);
    v16h bv11 = frag_ld(vb + 16 * VS, col16, half, VS, 32);
    v8f z = {};
    v8f o00 = wmma16(p01, bv01, wmma16(p00, bv00, z));
    v8f o01 = wmma16(p01, bv11, wmma16(p00, bv10, z));
    v8f o10 = wmma16(p11, bv01, wmma16(p10, bv00, z));
    v8f o11 = wmma16(p11, bv11, wmma16(p10, bv10, z));
#pragma unroll
    for (int r = 0; r < 8; ++r) {
      int r0 = mg * 32 + half * 8 + r;
      int r1 = r0 + 16;
      ol[r0 * XS + h * HD +  0 + col16] = (_Float16)o00[r];
      ol[r0 * XS + h * HD + 16 + col16] = (_Float16)o01[r];
      ol[r1 * XS + h * HD +  0 + col16] = (_Float16)o10[r];
      ol[r1 * XS + h * HD + 16 + col16] = (_Float16)o11[r];
    }
  }
  __syncthreads();

  // ---- stage 4: out = O @ proj_w + b ----
  {
    const int m0 = (wid >> 1) * 16;
    v16h a0 = frag_ld(ol + m0 * XS, col16, half, XS, 0);
    v16h a1 = frag_ld(ol + m0 * XS, col16, half, XS, 32);
    v16h a2 = frag_ld(ol + m0 * XS, col16, half, XS, 64);
    v16h a3 = frag_ld(ol + m0 * XS, col16, half, XS, 96);
    float* ob = out + (size_t)b * (NTOK * NDIM);
    for (int t = 0; t < 4; ++t) {
      int n0 = ((wid & 1) * 4 + t) * 16;
      const _Float16* prow = pT + (size_t)n0 * NDIM;
      v8f acc = {};
      acc = wmma16(a0, frag_ld(prow, col16, half, NDIM, 0),  acc);
      acc = wmma16(a1, frag_ld(prow, col16, half, NDIM, 32), acc);
      acc = wmma16(a2, frag_ld(prow, col16, half, NDIM, 64), acc);
      acc = wmma16(a3, frag_ld(prow, col16, half, NDIM, 96), acc);
      int c = n0 + col16;
      float pbv = proj_b[c];
#pragma unroll
      for (int r = 0; r < 8; ++r) {
        ob[(m0 + half * 8 + r) * NDIM + c] = acc[r] + pbv;
      }
    }
  }
}

extern "C" void kernel_launch(void* const* d_in, const int* in_sizes, int n_in,
                              void* d_out, int out_size, void* d_ws, size_t ws_size,
                              hipStream_t stream) {
  const float* x         = (const float*)d_in[0];
  const float* mask      = (const float*)d_in[1];
  const float* qkv_w     = (const float*)d_in[2];
  const float* qkv_b     = (const float*)d_in[3];
  const float* proj_w    = (const float*)d_in[4];
  const float* proj_b    = (const float*)d_in[5];
  const float* rel_table = (const float*)d_in[6];
  const int*   rel_index = (const int*)d_in[7];

  _Float16* wT    = (_Float16*)((char*)d_ws + WS_WT);
  _Float16* pT    = (_Float16*)((char*)d_ws + WS_PT);
  float*    biasw = (float*)((char*)d_ws + WS_BIAS);

  // 49152 + 16384 + 16384 elements = 320 blocks of 256
  wa_prep<<<320, 256, 0, stream>>>(qkv_w, proj_w, rel_table, rel_index, wT, pT, biasw);
  wa_main<<<NB, 256, 0, stream>>>(x, mask, qkv_b, proj_b, wT, pT, biasw, (float*)d_out);
}
